// Agent_14834817040470
// MI455X (gfx1250) — compile-verified, matrix operand order checked by reading, not run on hardware
//
#include <hip/hip_runtime.h>
#include <math.h>

// ---------------- problem constants ----------------
#define T_DIM   1024
#define B_DIM   256
#define OBS_DIM 64
#define ACT_N   8
#define H_DIM   256
#define TB      (T_DIM * B_DIM)          // 262144 policy rows
#define ROWS_V  ((T_DIM + 1) * B_DIM)    // 262400 value rows
#define LSTR    264                      // LDS row pitch (bf16) -> conflict-free b128 reads

#define LOG2PI_HALF 0.91893853320467274f
#define LN2         0.6931471805599453f
#define LOG_1P2     0.18232155679395463f
#define LOG_0P8    (-0.2231435513142097f)

typedef __attribute__((ext_vector_type(16))) __bf16 v16bf;
typedef __attribute__((ext_vector_type(8)))  __bf16 v8bf;
typedef __attribute__((ext_vector_type(4)))  __bf16 v4bf;
typedef __attribute__((ext_vector_type(8)))  float  v8f;
typedef __attribute__((ext_vector_type(4)))  float  v4f;

// ---------------- small device helpers ----------------
__device__ __forceinline__ __bf16 f2bf(float f) {
    unsigned u = __builtin_bit_cast(unsigned, f);
    unsigned r = (u + 0x7FFFu + ((u >> 16) & 1u)) >> 16;   // round-nearest-even
    unsigned short s = (unsigned short)r;
    return __builtin_bit_cast(__bf16, s);
}
__device__ __forceinline__ float bf2f(__bf16 b) {
    unsigned short s = __builtin_bit_cast(unsigned short, b);
    unsigned u = ((unsigned)s) << 16;
    return __builtin_bit_cast(float, u);
}
__device__ __forceinline__ float softplusf(float x) {
    return fmaxf(x, 0.f) + log1pf(__expf(-fabsf(x)));
}
__device__ __forceinline__ float siluf(float x) {
    // x * sigmoid(x) with hardware v_rcp_f32 (avoids IEEE div chain in hot epilogue)
    return x * __builtin_amdgcn_rcpf(1.f + __expf(-x));
}
__device__ __forceinline__ float ldjf(float x) {           // 2*(ln2 - x - softplus(-2x))
    return 2.f * (LN2 - x - softplusf(-2.f * x));
}
__device__ __forceinline__ float lp_term(float loc, float scale, float a) {
    float z = (a - loc) / scale;
    return -0.5f * z * z - (LOG2PI_HALF + logf(scale)) - ldjf(a);
}

// ---------------- weight fragment packing ----------------
// dst index = ((ntile*KC + kchunk)*32 + lane)*16 + e ; lane = (n&15) | (khalf<<4)
// k = kchunk*32 + (e>=8 ? 16 : 0) + khalf*8 + (e&7)   (matches 16-bit A/B WMMA layout)
__global__ void __launch_bounds__(256) k_pack_w(const float* __restrict__ src,
                                                __bf16* __restrict__ dst,
                                                int fout, int fin) {
    int total = fout * fin;
    int tid = blockIdx.x * 256 + threadIdx.x;
    if (tid >= total) return;
    int KC   = fin >> 5;
    int e    = tid & 15;
    int lane = (tid >> 4) & 31;
    int rest = tid >> 9;
    int kc   = rest % KC;
    int nt   = rest / KC;
    int n  = nt * 16 + (lane & 15);
    int kh = lane >> 4;
    int k  = kc * 32 + ((e & 8) ? 16 : 0) + kh * 8 + (e & 7);
    dst[tid] = f2bf(src[(size_t)n * fin + k]);
}

// ---------------- one WMMA layer: 32 rows, fin -> fout, SiLU, bf16 out to LDS ----------
__device__ __forceinline__ void wmma_layer(const __bf16* __restrict__ inL,
                                           __bf16* __restrict__ outL,
                                           const __bf16* __restrict__ wpack,
                                           const float* __restrict__ bias,
                                           int KC, int NT, int wave, int lane) {
    int n  = lane & 15;
    int kh = lane >> 4;
    int tiles = 2 * NT;                        // 2 row-tiles of 16
    for (int t = wave; t < tiles; t += 8) {
        int mt = t / NT;
        int nt = t - mt * NT;
        v8f acc = {0.f, 0.f, 0.f, 0.f, 0.f, 0.f, 0.f, 0.f};
        const __bf16* arow = inL + (size_t)(mt * 16 + n) * LSTR + kh * 8;
        const __bf16* bptr = wpack + ((size_t)(nt * KC) * 32 + lane) * 16;
        for (int kc = 0; kc < KC; ++kc) {
            v8bf lo = *(const v8bf*)(arow + kc * 32);
            v8bf hi = *(const v8bf*)(arow + kc * 32 + 16);
            v16bf a = __builtin_shufflevector(lo, hi, 0, 1, 2, 3, 4, 5, 6, 7,
                                              8, 9, 10, 11, 12, 13, 14, 15);
            v16bf b = *(const v16bf*)(bptr + (size_t)kc * 512);
            acc = __builtin_amdgcn_wmma_f32_16x16x32_bf16(false, a, false, b,
                                                          (short)0, acc, false, false);
        }
        float bia = bias[nt * 16 + n];
        int rowbase = mt * 16 + kh * 8;        // C/D: vgpr j -> row j + 8*(lane>>4)
#pragma unroll
        for (int j = 0; j < 8; ++j) {
            float o = acc[j] + bia;
            outL[(size_t)(rowbase + j) * LSTR + nt * 16 + n] = f2bf(siluf(o));
        }
    }
}

// head layer: 256 -> 16 logits, produce (loc | softplus(scale)+0.001) f32 tile [32][16]
__device__ __forceinline__ void wmma_head(const __bf16* __restrict__ inL,
                                          float* __restrict__ tile,
                                          const __bf16* __restrict__ wpack,
                                          const float* __restrict__ bias,
                                          int wave, int lane) {
    if (wave < 2) {                            // 2 tiles, KC = 8, NT = 1
        int n  = lane & 15;
        int kh = lane >> 4;
        int mt = wave;
        v8f acc = {0.f, 0.f, 0.f, 0.f, 0.f, 0.f, 0.f, 0.f};
        const __bf16* arow = inL + (size_t)(mt * 16 + n) * LSTR + kh * 8;
        const __bf16* bptr = wpack + (size_t)lane * 16;
        for (int kc = 0; kc < 8; ++kc) {
            v8bf lo = *(const v8bf*)(arow + kc * 32);
            v8bf hi = *(const v8bf*)(arow + kc * 32 + 16);
            v16bf a = __builtin_shufflevector(lo, hi, 0, 1, 2, 3, 4, 5, 6, 7,
                                              8, 9, 10, 11, 12, 13, 14, 15);
            v16bf b = *(const v16bf*)(bptr + (size_t)kc * 512);
            acc = __builtin_amdgcn_wmma_f32_16x16x32_bf16(false, a, false, b,
                                                          (short)0, acc, false, false);
        }
        float bia = bias[n];
        int rowbase = mt * 16 + kh * 8;
#pragma unroll
        for (int j = 0; j < 8; ++j) {
            float o = acc[j] + bia;
            float v = (n < 8) ? o : (softplusf(o) + 0.001f);
            tile[(rowbase + j) * 16 + n] = v;
        }
    }
}

// normalize 32x64 obs rows into LDS as bf16 (float4 loads, b64 LDS stores)
__device__ __forceinline__ void load_norm_tile(const float* __restrict__ src, size_t row0,
                                               const float* s_rm, const float* s_inv,
                                               __bf16* __restrict__ buf, int tid) {
#pragma unroll
    for (int j = 0; j < 2; ++j) {
        int q = tid + j * 256;                 // 512 float4 groups
        int r = q >> 4, f4 = q & 15;
        v4f x = *(const v4f*)(src + (row0 + r) * 64 + f4 * 4);
        v4bf o;
#pragma unroll
        for (int c = 0; c < 4; ++c) {
            float xn = fminf(fmaxf((x[c] - s_rm[f4 * 4 + c]) * s_inv[f4 * 4 + c], -5.f), 5.f);
            o[c] = f2bf(xn);
        }
        *(v4bf*)(buf + (size_t)r * LSTR + f4 * 4) = o;
    }
}

// ---------------- fused guider+learner policy MLP + value MLP + per-row stats ------
__global__ void __launch_bounds__(256) k_fused(
    const float* __restrict__ obs, const float* __restrict__ pobs,
    const float* __restrict__ rm, const float* __restrict__ rv,
    const __bf16* __restrict__ w0, const float* __restrict__ b0,
    const __bf16* __restrict__ w1, const float* __restrict__ b1,
    const __bf16* __restrict__ w2, const float* __restrict__ b2,
    const __bf16* __restrict__ vw0p, const float* __restrict__ vb0,
    const __bf16* __restrict__ vw1p, const float* __restrict__ vb1,
    const float* __restrict__ vw2, const float* __restrict__ vb2,
    const float* __restrict__ action, const float* __restrict__ logits,
    const float* __restrict__ noise,
    float* __restrict__ glp, float* __restrict__ llp, float* __restrict__ blp,
    float* __restrict__ baseline,
    float* __restrict__ pkl, float* __restrict__ pklg, float* __restrict__ pent) {

    __shared__ __align__(16) __bf16 bufA[32 * LSTR];
    __shared__ __align__(16) __bf16 bufB[32 * LSTR];
    __shared__ __align__(16) __bf16 bufC[32 * LSTR];
    __shared__ float gtile[32 * 16];
    __shared__ float ltile[32 * 16];
    __shared__ float s_rm[64], s_inv[64];

    int tid = threadIdx.x;
    int wave = tid >> 5, lane = tid & 31;
    size_t row0 = (size_t)blockIdx.x * 32;

    if (tid < 64) {
        float var = rv[tid] * (float)(1.0 / 1000001.0);     // rv / (NUM_STEPS+1)
        var = fminf(fmaxf(var, 1e-6f), 1e6f);
        s_rm[tid]  = rm[tid];
        s_inv[tid] = rsqrtf(var);
    }
    __syncthreads();

    // ---- pass 0: observation -> guider logits + value baseline ----
    load_norm_tile(obs, row0, s_rm, s_inv, bufA, tid);
    __syncthreads();
    wmma_layer(bufA, bufB, w0, b0, 2, 16, wave, lane);      // policy L0
    wmma_layer(bufA, bufC, vw0p, vb0, 2, 16, wave, lane);   // value  L0 (same phase: reads bufA)
    __syncthreads();
    wmma_layer(bufB, bufA, w1, b1, 8, 16, wave, lane);      // policy L1
    __syncthreads();
    wmma_layer(bufC, bufB, vw1p, vb1, 8, 16, wave, lane);   // value  L1
    __syncthreads();
    wmma_head(bufA, gtile, w2, b2, wave, lane);             // policy head -> gtile
    {                                                       // value head: 8 threads/row
        int r = tid >> 3, p = tid & 7;
        float s = 0.f;
        const __bf16* rp = bufB + (size_t)r * LSTR + p * 32;
        const float* wv = vw2 + p * 32;
#pragma unroll
        for (int k = 0; k < 32; ++k) s += bf2f(rp[k]) * wv[k];
        s += __shfl_xor(s, 1, 32);
        s += __shfl_xor(s, 2, 32);
        s += __shfl_xor(s, 4, 32);
        if (p == 0) baseline[row0 + r] = s + vb2[0];
    }
    __syncthreads();

    // ---- pass 1: pobservation -> learner logits ----
    load_norm_tile(pobs, row0, s_rm, s_inv, bufA, tid);
    __syncthreads();
    wmma_layer(bufA, bufB, w0, b0, 2, 16, wave, lane);
    __syncthreads();
    wmma_layer(bufB, bufA, w1, b1, 8, 16, wave, lane);
    __syncthreads();
    wmma_head(bufA, ltile, w2, b2, wave, lane);
    __syncthreads();

    // ---- per-row stats ----
    float kl_s = 0.f, klg_s = 0.f, ent_s = 0.f;
    if (tid < 32) {
        size_t grow = row0 + tid;
        const float* ap = action + grow * 8;
        const float* np = noise + grow * 8;
        const float* lg = logits + grow * 16;
        float blp_v = 0.f, glp_v = 0.f, llp_v = 0.f, kl = 0.f, ent = 0.f;
#pragma unroll
        for (int d = 0; d < 8; ++d) {
            float a  = ap[d];
            float bl = lg[d];
            float bs = softplusf(lg[8 + d]) + 0.001f;
            blp_v += lp_term(bl, bs, a);
            float gl = gtile[tid * 16 + d], gs = gtile[tid * 16 + 8 + d];
            glp_v += lp_term(gl, gs, a);
            float ll = ltile[tid * 16 + d], ls = ltile[tid * 16 + 8 + d];
            llp_v += lp_term(ll, ls, a);
            float sr = gs / ls;  float vr = sr * sr;
            float dm = (gl - ll) / ls;
            kl += 0.5f * (vr + dm * dm - 1.f - logf(vr));
            float dist = bl + bs * np[d];
            ent += (0.5f + LOG2PI_HALF + logf(bs)) + ldjf(dist);
        }
        kl *= 0.125f;                                       // mean over ACT dims
        float diff = glp_v - llp_v;
        float mk = (diff > LOG_1P2 || diff < LOG_0P8) ? 1.f : 0.f;
        glp[grow] = glp_v; llp[grow] = llp_v; blp[grow] = blp_v;
        kl_s = kl; klg_s = kl * mk; ent_s = ent;
    }
    if (wave == 0) {                                        // deterministic block partials
        for (int o = 16; o > 0; o >>= 1) {
            kl_s  += __shfl_xor(kl_s,  o, 32);
            klg_s += __shfl_xor(klg_s, o, 32);
            ent_s += __shfl_xor(ent_s, o, 32);
        }
        if (lane == 0) {
            pkl[blockIdx.x]  = kl_s;
            pklg[blockIdx.x] = klg_s;
            pent[blockIdx.x] = ent_s;
        }
    }
}

// ---------------- value MLP tail (bootstrap rows TB .. TB+B-1) ----------------------
__global__ void __launch_bounds__(256) k_value_tail(
    const float* __restrict__ obs, size_t row_base,
    const float* __restrict__ rm, const float* __restrict__ rv,
    const __bf16* __restrict__ w0, const float* __restrict__ b0,
    const __bf16* __restrict__ w1, const float* __restrict__ b1,
    const float* __restrict__ vw2, const float* __restrict__ vb2,
    float* __restrict__ baseline) {

    __shared__ __align__(16) __bf16 bufA[32 * LSTR];
    __shared__ __align__(16) __bf16 bufB[32 * LSTR];
    __shared__ float s_rm[64], s_inv[64];

    int tid = threadIdx.x;
    int wave = tid >> 5, lane = tid & 31;
    size_t row0 = row_base + (size_t)blockIdx.x * 32;

    if (tid < 64) {
        float var = rv[tid] * (float)(1.0 / 1000001.0);
        var = fminf(fmaxf(var, 1e-6f), 1e6f);
        s_rm[tid]  = rm[tid];
        s_inv[tid] = rsqrtf(var);
    }
    __syncthreads();

    load_norm_tile(obs, row0, s_rm, s_inv, bufA, tid);
    __syncthreads();
    wmma_layer(bufA, bufB, w0, b0, 2, 16, wave, lane);
    __syncthreads();
    wmma_layer(bufB, bufA, w1, b1, 8, 16, wave, lane);
    __syncthreads();

    {
        int r = tid >> 3, p = tid & 7;
        float s = 0.f;
        const __bf16* rp = bufA + (size_t)r * LSTR + p * 32;
        const float* wv = vw2 + p * 32;
#pragma unroll
        for (int k = 0; k < 32; ++k) s += bf2f(rp[k]) * wv[k];
        s += __shfl_xor(s, 1, 32);
        s += __shfl_xor(s, 2, 32);
        s += __shfl_xor(s, 4, 32);
        if (p == 0) baseline[row0 + r] = s + vb2[0];
    }
}

// ---------------- GAE reverse scan (one column per thread) -------------------------
__global__ void __launch_bounds__(256) k_gae(
    const float* __restrict__ reward, const float* __restrict__ done,
    const float* __restrict__ trunc, const float* __restrict__ baseline,
    float* __restrict__ adv, double* __restrict__ scal) {

    int b = threadIdx.x;
    double s_adv = 0.0, s_adv2 = 0.0, s_vsq = 0.0;
    float acc = 0.f;
    float vsn = baseline[(size_t)T_DIM * B_DIM + b];        // bootstrap
    for (int t = T_DIM - 1; t >= 0; --t) {
        size_t i = (size_t)t * B_DIM + b;
        float r  = reward[i];
        float tm = 1.f - trunc[i];
        float term = done[i] * tm;
        float vt  = baseline[i];
        float vt1 = baseline[i + B_DIM];
        float g1  = 0.97f * (1.f - term);
        float delta = (r + g1 * vt1 - vt) * tm;
        acc = delta + g1 * tm * 0.95f * acc;
        float vs_t = acc + vt;
        float a = (r + g1 * vsn - vt) * tm;
        adv[i] = a;
        float verr = vs_t - vt;
        s_vsq  += (double)(verr * verr);
        s_adv  += (double)a;
        s_adv2 += (double)a * (double)a;
        vsn = vs_t;
    }
    __shared__ double sh[3 * 256];
    sh[b] = s_adv; sh[256 + b] = s_adv2; sh[512 + b] = s_vsq;
    __syncthreads();
    if (b == 0) {
        double a = 0, a2 = 0, v = 0;
        for (int i = 0; i < 256; ++i) { a += sh[i]; a2 += sh[256 + i]; v += sh[512 + i]; }
        scal[0] = a; scal[1] = a2; scal[2] = v;
    }
}

// ---------------- policy losses (needs adv mean/std) -------------------------------
__global__ void __launch_bounds__(256) k_ploss(
    const float* __restrict__ adv, const float* __restrict__ glp,
    const float* __restrict__ llp, const float* __restrict__ blp,
    const double* __restrict__ scal,
    float* __restrict__ pplg, float* __restrict__ ppll) {

    const double N = (double)TB;
    double mean = scal[0] / N;
    double var  = (scal[1] - scal[0] * scal[0] / N) / (N - 1.0);
    float denom = (float)sqrt(fmax(var, 0.0)) + 1e-5f;
    float fmean = (float)mean;

    int idx = blockIdx.x * 256 + threadIdx.x;
    float a  = (adv[idx] - fmean) / denom;
    float g  = glp[idx], l = llp[idx], bb = blp[idx];

    float rho = __expf(g - bb);
    float c   = fminf(fmaxf(g - l, LOG_0P8), LOG_1P2);
    float rc  = fminf(fmaxf(__expf(c + l - bb), 0.7f), 1.3f);
    float gterm = fminf(rho * a, rc * a);

    float rho2 = __expf(fminf(l - bb, 10.f));
    float r2c  = fminf(fmaxf(rho2, 0.7f), 1.3f);
    float lterm = fminf(rho2 * a, r2c * a);

    int wave = threadIdx.x >> 5, lane = threadIdx.x & 31;
    for (int o = 16; o > 0; o >>= 1) {
        gterm += __shfl_xor(gterm, o, 32);
        lterm += __shfl_xor(lterm, o, 32);
    }
    __shared__ float shg[8], shl[8];
    if (lane == 0) { shg[wave] = gterm; shl[wave] = lterm; }
    __syncthreads();
    if (threadIdx.x == 0) {
        float sg = 0.f, sl = 0.f;
        for (int i = 0; i < 8; ++i) { sg += shg[i]; sl += shl[i]; }
        pplg[blockIdx.x] = sg;
        ppll[blockIdx.x] = sl;
    }
}

// ---------------- deterministic reduction of partials ------------------------------
__global__ void __launch_bounds__(256) k_reduce(const float* __restrict__ in, int n,
                                                double* __restrict__ out) {
    double s = 0.0;
    for (int i = threadIdx.x; i < n; i += 256) s += (double)in[i];
    __shared__ double sh[256];
    sh[threadIdx.x] = s;
    __syncthreads();
    if (threadIdx.x == 0) {
        double t = 0.0;
        for (int i = 0; i < 256; ++i) t += sh[i];
        *out = t;
    }
}

// ---------------- finalize 4 output scalars ----------------------------------------
__global__ void k_finalize(const double* __restrict__ scal, float* __restrict__ out) {
    if (threadIdx.x == 0 && blockIdx.x == 0) {
        const double N = (double)TB;
        double kl_learner = scal[3] / N;
        double kl_guider  = scal[4] / N;
        double ent        = scal[5] / N;
        double gpl = -(scal[6] / N);
        double lpl = -(scal[7] / N);
        double v_loss   = (scal[2] / N) * 0.25;
        double ent_loss = -0.01 * ent;
        double learner_loss = kl_learner + lpl;             // ALPHA = 1
        double guider_loss  = kl_guider + gpl;
        double total = learner_loss + guider_loss + v_loss + ent_loss;
        out[0] = (float)total;
        out[1] = (float)guider_loss;
        out[2] = (float)learner_loss;
        out[3] = (float)kl_learner;
    }
}

// ---------------- launcher ---------------------------------------------------------
extern "C" void kernel_launch(void* const* d_in, const int* in_sizes, int n_in,
                              void* d_out, int out_size, void* d_ws, size_t ws_size,
                              hipStream_t stream) {
    (void)in_sizes; (void)n_in; (void)out_size; (void)ws_size;
    const float* observation  = (const float*)d_in[0];
    const float* pobservation = (const float*)d_in[1];
    const float* reward  = (const float*)d_in[2];
    const float* done    = (const float*)d_in[3];
    const float* trunc   = (const float*)d_in[4];
    const float* logits  = (const float*)d_in[5];
    const float* action  = (const float*)d_in[6];
    const float* noise   = (const float*)d_in[7];
    const float* pw0 = (const float*)d_in[8];  const float* pb0 = (const float*)d_in[9];
    const float* pw1 = (const float*)d_in[10]; const float* pb1 = (const float*)d_in[11];
    const float* pw2 = (const float*)d_in[12]; const float* pb2 = (const float*)d_in[13];
    const float* vw0 = (const float*)d_in[14]; const float* vb0 = (const float*)d_in[15];
    const float* vw1 = (const float*)d_in[16]; const float* vb1 = (const float*)d_in[17];
    const float* vw2 = (const float*)d_in[18]; const float* vb2 = (const float*)d_in[19];
    const float* rm  = (const float*)d_in[20]; const float* rv  = (const float*)d_in[21];

    char* ws = (char*)d_ws;
    size_t off = 0;
    auto take = [&](size_t bytes) -> char* {
        char* p = ws + off;
        off = (off + bytes + 255) & ~(size_t)255;
        return p;
    };
    double* scal  = (double*)take(8 * sizeof(double));
    __bf16* w0p   = (__bf16*)take((size_t)H_DIM * OBS_DIM * 2);
    __bf16* w1p   = (__bf16*)take((size_t)H_DIM * H_DIM * 2);
    __bf16* w2p   = (__bf16*)take((size_t)16 * H_DIM * 2);
    __bf16* vw0p  = (__bf16*)take((size_t)H_DIM * OBS_DIM * 2);
    __bf16* vw1p  = (__bf16*)take((size_t)H_DIM * H_DIM * 2);
    float* glp      = (float*)take((size_t)TB * 4);
    float* llp      = (float*)take((size_t)TB * 4);
    float* blp      = (float*)take((size_t)TB * 4);
    float* baseline = (float*)take((size_t)ROWS_V * 4);
    float* adv      = (float*)take((size_t)TB * 4);
    float* pkl      = (float*)take((size_t)(TB / 32) * 4);
    float* pklg     = (float*)take((size_t)(TB / 32) * 4);
    float* pent     = (float*)take((size_t)(TB / 32) * 4);
    float* pplg     = (float*)take((size_t)(TB / 256) * 4);
    float* ppll     = (float*)take((size_t)(TB / 256) * 4);

    // 1) pack weights into WMMA B-fragment layout (bf16)
    auto pack = [&](const float* s, __bf16* d, int fo, int fi) {
        int total = fo * fi;
        k_pack_w<<<(total + 255) / 256, 256, 0, stream>>>(s, d, fo, fi);
    };
    pack(pw0, w0p, 256, 64);
    pack(pw1, w1p, 256, 256);
    pack(pw2, w2p, 16, 256);
    pack(vw0, vw0p, 256, 64);
    pack(vw1, vw1p, 256, 256);

    // 2) fused guider+learner policy MLPs + value MLP (obs read exactly once)
    k_fused<<<TB / 32, 256, 0, stream>>>(observation, pobservation, rm, rv,
                                         w0p, pb0, w1p, pb1, w2p, pb2,
                                         vw0p, vb0, vw1p, vb1, vw2, vb2,
                                         action, logits, noise,
                                         glp, llp, blp, baseline,
                                         pkl, pklg, pent);

    // 3) value MLP tail: bootstrap rows [TB, TB+B)
    k_value_tail<<<B_DIM / 32, 256, 0, stream>>>(observation, (size_t)TB, rm, rv,
                                                 vw0p, vb0, vw1p, vb1, vw2, vb2, baseline);

    // 4) GAE reverse scan -> adv + adv/vloss sums
    k_gae<<<1, 256, 0, stream>>>(reward, done, trunc, baseline, adv, scal);

    // 5) deterministic reductions of MLP-stage partials
    k_reduce<<<1, 256, 0, stream>>>(pkl,  TB / 32, scal + 3);
    k_reduce<<<1, 256, 0, stream>>>(pklg, TB / 32, scal + 4);
    k_reduce<<<1, 256, 0, stream>>>(pent, TB / 32, scal + 5);

    // 6) policy losses (need adv mean/std) then reduce
    k_ploss<<<TB / 256, 256, 0, stream>>>(adv, glp, llp, blp, scal, pplg, ppll);
    k_reduce<<<1, 256, 0, stream>>>(pplg, TB / 256, scal + 6);
    k_reduce<<<1, 256, 0, stream>>>(ppll, TB / 256, scal + 7);

    // 7) combine into the 4 output scalars
    k_finalize<<<1, 64, 0, stream>>>(scal, (float*)d_out);
}
